// VggModel_40879498728834
// MI455X (gfx1250) — compile-verified, matrix operand order checked by reading, not run on hardware
//
#include <hip/hip_runtime.h>
#include <hip/hip_bf16.h>

// ---------------------------------------------------------------------------
// VggModel forward, fused into 3 WMMA GEMM kernels for gfx1250 (MI455X).
// Memory-bound (~100MB streamed, 23.6 GFLOP => ~4.3us floor at 23.3 TB/s).
// bf16 WMMA (f32 accumulate); dots[512,2000] never hits HBM (brand-weighted
// mean fused into GEMM3 epilogue). Double-buffered LDS with register staging:
// next tile's global_load_b128s are issued before the current tile's WMMAs so
// HBM latency hides under ds_load_b128 + v_wmma work. fp32->bf16 packing via
// v_perm_b32.
// ---------------------------------------------------------------------------

typedef __attribute__((ext_vector_type(16))) __bf16        v16bf;
typedef __attribute__((ext_vector_type(8)))  float         v8f;
typedef __attribute__((ext_vector_type(8)))  unsigned int  v8u;

#define BM 128
#define BN 128
#define BK 32
#define THREADS 256
#define STAGE_ITERS 4   // (BM*BK/4) float4 loads / THREADS == 4, same for B

// Pack two fp32 into one bf16x2 dword: round bias + v_perm_b32 byte select.
__device__ __forceinline__ unsigned int pack_bf16x2(float a, float b) {
    const unsigned int ua = __builtin_bit_cast(unsigned int, a) + 0x8000u;
    const unsigned int ub = __builtin_bit_cast(unsigned int, b) + 0x8000u;
    return __builtin_amdgcn_perm(ub, ua, 0x07060302u);
}

__device__ __forceinline__ unsigned short f2bf(float f) {
    return (unsigned short)((__builtin_bit_cast(unsigned int, f) + 0x8000u) >> 16);
}

// Load one 16x32 bf16 fragment from an LDS tile stored row-major [rows][BK].
// Per ISA 7.12.2: lanes 0-15 hold row=lane, K in {0..7,16..23}; lanes 16-31
// hold row=lane-16, K in {8..15,24..31}  => two ds_load_b128 per lane.
__device__ __forceinline__ v16bf load_frag(const unsigned short* tile, int row, int lane) {
    const int kOff = (lane < 16) ? 0 : 8;
    const uint4 lo = *(const uint4*)(tile + row * BK + kOff);
    const uint4 hi = *(const uint4*)(tile + row * BK + kOff + 16);
    v8u r;
    r[0] = lo.x; r[1] = lo.y; r[2] = lo.z; r[3] = lo.w;
    r[4] = hi.x; r[5] = hi.y; r[6] = hi.z; r[7] = hi.w;
    return __builtin_bit_cast(v16bf, r);
}

// B_IS_NK: 0 => B is [K,N] row-major (W1, W2); 1 => B is [N,K] row-major (aspects)
// EPI:     0 => C = leaky_relu(acc + bias)        (GEMM1 -> h)
//          1 => C = acc + bias                    (GEMM2 -> im_ft)
//          2 => out[m] += mean_n(brand_w * acc)   (GEMM3, fused reduce)
template <int B_IS_NK, int EPI>
__global__ __launch_bounds__(THREADS) void gemm_wmma(
    const float* __restrict__ A, const float* __restrict__ B,
    const float* __restrict__ bias, float* __restrict__ C,
    int M, int N, int K,
    const long long* __restrict__ brand,
    const float* __restrict__ brand_emb,
    float* __restrict__ out)
{
    __shared__ __align__(16) unsigned short As[2][BM * BK];
    __shared__ __align__(16) unsigned short Bs[2][BN * BK];
    __shared__ float rowsum[BM];

    const int tid   = threadIdx.x;
    const int lane  = tid & 31;
    const int wave  = tid >> 5;       // 8 wave32s
    const int waveM = wave & 1;       // 2 waves along M (64 rows each)
    const int waveN = wave >> 1;      // 4 waves along N (32 cols each)
    const int bM    = blockIdx.x * BM;
    const int bN    = blockIdx.y * BN;

    if (EPI == 2 && tid < BM) rowsum[tid] = 0.0f;

    float4 aReg[STAGE_ITERS];
    float4 bReg[STAGE_ITERS];

    // Issue the global loads for the tile at K-offset k0 (no waits here:
    // up to 8 global_load_b128 in flight per thread).
    auto issue_loads = [&](int k0) {
        #pragma unroll
        for (int u = 0; u < STAGE_ITERS; ++u) {
            const int t   = tid + u * THREADS;
            const int row = t >> 3;               // 8 float4 per A row
            const int cv  = t & 7;
            aReg[u] = *(const float4*)(A + (size_t)(bM + row) * K + k0 + cv * 4);
        }
        if (B_IS_NK) {
            #pragma unroll
            for (int u = 0; u < STAGE_ITERS; ++u) {
                const int t   = tid + u * THREADS;
                const int row = t >> 3;           // n within tile
                const int cv  = t & 7;
                const int gn  = bN + row;
                float4 v = make_float4(0.f, 0.f, 0.f, 0.f);
                if (gn < N) v = *(const float4*)(B + (size_t)gn * K + k0 + cv * 4);
                bReg[u] = v;
            }
        } else {
            #pragma unroll
            for (int u = 0; u < STAGE_ITERS; ++u) {
                const int t    = tid + u * THREADS;
                const int krow = t >> 5;          // 32 float4 per K-row
                const int cv   = t & 31;
                bReg[u] = *(const float4*)(B + (size_t)(k0 + krow) * N + bN + cv * 4);
            }
        }
    };

    // Convert the staged registers to bf16 and commit them to one LDS buffer.
    auto commit_lds = [&](unsigned short* As_, unsigned short* Bs_) {
        #pragma unroll
        for (int u = 0; u < STAGE_ITERS; ++u) {
            const int t   = tid + u * THREADS;
            const int row = t >> 3;
            const int cv  = t & 7;
            *(uint2*)(As_ + row * BK + cv * 4) =
                make_uint2(pack_bf16x2(aReg[u].x, aReg[u].y),
                           pack_bf16x2(aReg[u].z, aReg[u].w));
        }
        if (B_IS_NK) {
            #pragma unroll
            for (int u = 0; u < STAGE_ITERS; ++u) {
                const int t   = tid + u * THREADS;
                const int row = t >> 3;
                const int cv  = t & 7;
                *(uint2*)(Bs_ + row * BK + cv * 4) =
                    make_uint2(pack_bf16x2(bReg[u].x, bReg[u].y),
                               pack_bf16x2(bReg[u].z, bReg[u].w));
            }
        } else {
            #pragma unroll
            for (int u = 0; u < STAGE_ITERS; ++u) {
                const int t    = tid + u * THREADS;
                const int krow = t >> 5;
                const int cv   = t & 31;
                Bs_[(cv * 4 + 0) * BK + krow] = f2bf(bReg[u].x);
                Bs_[(cv * 4 + 1) * BK + krow] = f2bf(bReg[u].y);
                Bs_[(cv * 4 + 2) * BK + krow] = f2bf(bReg[u].z);
                Bs_[(cv * 4 + 3) * BK + krow] = f2bf(bReg[u].w);
            }
        }
    };

    v8f acc[4][2];
    #pragma unroll
    for (int i = 0; i < 4; ++i)
        #pragma unroll
        for (int j = 0; j < 2; ++j) { v8f z = {}; acc[i][j] = z; }

    // Prologue: stage tile 0 into buffer 0.
    issue_loads(0);
    commit_lds(As[0], Bs[0]);
    __syncthreads();

    const int nK = K / BK;
    for (int kt = 0; kt < nK; ++kt) {
        const int buf = kt & 1;

        // Issue next tile's global loads before this tile's WMMAs.
        if (kt + 1 < nK) {
            issue_loads((kt + 1) * BK);
            // Prefetch tile kt+2 into cache (global_prefetch_b8).
            if (kt + 2 < nK) {
                __builtin_prefetch(A + (size_t)(bM + (tid >> 1)) * K + (kt + 2) * BK, 0, 3);
                if (B_IS_NK) {
                    const int gn = bN + (tid >> 1);
                    if (gn < N) __builtin_prefetch(B + (size_t)gn * K + (kt + 2) * BK, 0, 3);
                } else {
                    __builtin_prefetch(B + (size_t)((kt + 2) * BK + (tid >> 3)) * N + bN, 0, 3);
                }
            }
        }

        // Compute on the current buffer: 4x2 v_wmma_f32_16x16x32_bf16 per wave.
        v16bf afrag[4];
        #pragma unroll
        for (int i = 0; i < 4; ++i)
            afrag[i] = load_frag(As[buf], waveM * 64 + i * 16 + (lane & 15), lane);
        v16bf bfrag[2];
        #pragma unroll
        for (int j = 0; j < 2; ++j)
            bfrag[j] = load_frag(Bs[buf], waveN * 32 + j * 16 + (lane & 15), lane);

        #pragma unroll
        for (int i = 0; i < 4; ++i)
            #pragma unroll
            for (int j = 0; j < 2; ++j)
                acc[i][j] = __builtin_amdgcn_wmma_f32_16x16x32_bf16(
                    false, afrag[i], false, bfrag[j],
                    (short)0, acc[i][j], false, false);

        // Commit next tile into the other buffer (waits the global loads here,
        // after the WMMAs have been issued).
        if (kt + 1 < nK) commit_lds(As[buf ^ 1], Bs[buf ^ 1]);
        __syncthreads();
    }

    // ---- epilogues ----
    if (EPI == 0 || EPI == 1) {
        #pragma unroll
        for (int i = 0; i < 4; ++i) {
            #pragma unroll
            for (int j = 0; j < 2; ++j) {
                const int gm0 = bM + waveM * 64 + i * 16 + ((lane < 16) ? 0 : 8);
                const int gn  = bN + waveN * 32 + j * 16 + (lane & 15);
                const float bv = bias[gn];
                #pragma unroll
                for (int r = 0; r < 8; ++r) {
                    float v = acc[i][j][r] + bv;
                    if (EPI == 0) v = (v > 0.0f) ? v : 0.01f * v;
                    C[(size_t)(gm0 + r) * N + gn] = v;
                }
            }
        }
    } else {
        // Fused: out[m] += (1/2000) * sum_n brand_emb[brand[m],n] * dots[m,n]
        #pragma unroll
        for (int i = 0; i < 4; ++i) {
            #pragma unroll
            for (int r = 0; r < 8; ++r) {
                const int mLocal = waveM * 64 + i * 16 + r + ((lane < 16) ? 0 : 8);
                const long long bidx = brand[bM + mLocal];
                float s = 0.0f;
                #pragma unroll
                for (int j = 0; j < 2; ++j) {
                    const int gn = bN + waveN * 32 + j * 16 + (lane & 15);
                    const float w = (gn < N) ? brand_emb[(size_t)bidx * N + gn] : 0.0f;
                    s += w * acc[i][j][r];
                }
                atomicAdd(&rowsum[mLocal], s);   // ds_add_f32
            }
        }
        __syncthreads();
        if (tid < BM)
            atomicAdd(&out[bM + tid], rowsum[tid] * (1.0f / 2000.0f)); // global_atomic_add_f32
    }
}

__global__ void zero_kernel(float* __restrict__ p, int n) {
    const int i = blockIdx.x * blockDim.x + threadIdx.x;
    if (i < n) p[i] = 0.0f;
}

extern "C" void kernel_launch(void* const* d_in, const int* in_sizes, int n_in,
                              void* d_out, int out_size, void* d_ws, size_t ws_size,
                              hipStream_t stream) {
    const float*     image     = (const float*)d_in[0];      // [512, 4096]
    const long long* brand     = (const long long*)d_in[1];  // [512] int64
    const float*     W1        = (const float*)d_in[2];      // [4096, 4096]
    const float*     b1        = (const float*)d_in[3];      // [4096]
    const float*     W2        = (const float*)d_in[4];      // [4096, 1024]
    const float*     b2        = (const float*)d_in[5];      // [1024]
    const float*     brand_emb = (const float*)d_in[6];      // [10000, 2000]
    const float*     aspects   = (const float*)d_in[7];      // [2000, 1024]
    float*           out       = (float*)d_out;              // [512, 1]

    float* h     = (float*)d_ws;          // [512, 4096] scratch
    float* im_ft = h + 512 * 4096;        // [512, 1024] scratch

    const dim3 blk(THREADS);

    // GEMM1: h = leaky_relu(image @ W1 + b1)
    gemm_wmma<0, 0><<<dim3(512 / BM, 4096 / BN), blk, 0, stream>>>(
        image, W1, b1, h, 512, 4096, 4096, nullptr, nullptr, nullptr);

    // GEMM2: im_ft = h @ W2 + b2
    gemm_wmma<0, 1><<<dim3(512 / BM, 1024 / BN), blk, 0, stream>>>(
        h, W2, b2, im_ft, 512, 1024, 4096, nullptr, nullptr, nullptr);

    // out = 0 (harness poisons d_out; we accumulate with atomics)
    zero_kernel<<<(512 + 255) / 256, 256, 0, stream>>>(out, 512);

    // GEMM3 fused: dots = im_ft @ aspects^T; out = mean(brand_w * dots, axis=1)
    gemm_wmma<1, 2><<<dim3(512 / BM, (2000 + BN - 1) / BN), blk, 0, stream>>>(
        im_ft, aspects, nullptr, nullptr, 512, 2000, 1024, brand, brand_emb, out);
}
